// TransformerBlock_19971597926574
// MI455X (gfx1250) — compile-verified, matrix operand order checked by reading, not run on hardware
//
#include <hip/hip_runtime.h>
#include <hip/hip_bf16.h>

// ---------------- problem constants ----------------
#define BB   1024
#define TT   128
#define DD   128
#define HH   4
#define HDIM 32
#define FF   512
#define MROWS (BB*TT)            // 131072
#define SCALE 0.08838834764831845f   // 128^-0.5
#define LNEPS 1e-5f

typedef __attribute__((ext_vector_type(8)))  _Float16 v8h;
typedef __attribute__((ext_vector_type(16))) _Float16 v16h;
typedef __attribute__((ext_vector_type(4)))  _Float16 v4h;
typedef __attribute__((ext_vector_type(8)))  float    v8f;

// CDNA5 16-bit A/B fragment: lane l holds row (l&15); K elements are two
// contiguous 8-half runs: K = khalf+e (e<8) and 16+khalf+(e-8), khalf = (l&16)?8:0.
// => two 16-byte loads per lane per 16x16x32 WMMA operand.
__device__ __forceinline__ v16h load_frag(const _Float16* __restrict__ base,
                                          int stride, int row, int k0, int khalf) {
    const _Float16* p = base + (size_t)row * stride + k0 + khalf;
    v8h lo = *(const v8h*)(p);
    v8h hi = *(const v8h*)(p + 16);
    v16h r;
#pragma unroll
    for (int i = 0; i < 8; ++i) { r[i] = lo[i]; r[8 + i] = hi[i]; }
    return r;
}

// ---------------- weight fp32 -> f16 pack ----------------
// layout in ws: [0..49151]=Wq|Wk|Wv (as [384,128]), [49152..65535]=Wo,
// [65536..131071]=W1, [131072..196607]=W2
__global__ __launch_bounds__(256) void cvt_weights_kernel(
    const float* __restrict__ Wq, const float* __restrict__ Wk,
    const float* __restrict__ Wv, const float* __restrict__ Wo,
    const float* __restrict__ W1, const float* __restrict__ W2,
    _Float16* __restrict__ out) {
    int i = blockIdx.x * 256 + threadIdx.x;
    float v;
    if      (i <  16384) v = Wq[i];
    else if (i <  32768) v = Wk[i - 16384];
    else if (i <  49152) v = Wv[i - 32768];
    else if (i <  65536) v = Wo[i - 49152];
    else if (i < 131072) v = W1[i - 65536];
    else                 v = W2[i - 131072];
    out[i] = (_Float16)v;
}

// ---------------- LayerNorm (fp32 in, f16 out), one wave32 per row ----------------
__global__ __launch_bounds__(256) void ln_f16_kernel(
    const float* __restrict__ x, const float* __restrict__ g,
    const float* __restrict__ b, _Float16* __restrict__ out) {
    int lane = threadIdx.x & 31;
    int row  = blockIdx.x * 8 + (threadIdx.x >> 5);
    const float4* xr = (const float4*)(x + (size_t)row * DD);
    float4 v = xr[lane];                       // 32 lanes * 4 = 128
    float s = v.x + v.y + v.z + v.w;
#pragma unroll
    for (int off = 16; off >= 1; off >>= 1) s += __shfl_xor(s, off, 32);
    float mu = s * (1.0f / DD);
    float dx = v.x - mu, dy = v.y - mu, dz = v.z - mu, dw = v.w - mu;
    float q = dx * dx + dy * dy + dz * dz + dw * dw;
#pragma unroll
    for (int off = 16; off >= 1; off >>= 1) q += __shfl_xor(q, off, 32);
    float rstd = rsqrtf(q * (1.0f / DD) + LNEPS);
    float4 gv = ((const float4*)g)[lane];
    float4 bv = ((const float4*)b)[lane];
    v4h o;
    o[0] = (_Float16)(dx * rstd * gv.x + bv.x);
    o[1] = (_Float16)(dy * rstd * gv.y + bv.y);
    o[2] = (_Float16)(dz * rstd * gv.z + bv.z);
    o[3] = (_Float16)(dw * rstd * gv.w + bv.w);
    *(v4h*)(out + (size_t)row * DD + lane * 4) = o;
}

// ---------------- generic WMMA GEMM: C[M,N] = A[M,K] @ W[N,K]^T ----------------
// 256 threads = 8 waves, block tile 32(M) x 64(N); each wave one 16x16 WMMA tile.
// Optional bias (fp32), ReLU, fp32 residual; outputs fp32 and/or f16.
__global__ __launch_bounds__(256) void wmma_gemm_kernel(
    const _Float16* __restrict__ A, int lda,
    const _Float16* __restrict__ W, int ldw,
    const float* __restrict__ bias, const float* __restrict__ res,
    float* __restrict__ outF, _Float16* __restrict__ outH,
    int N, int K, int relu) {
    int lane  = threadIdx.x & 31;
    int wave  = threadIdx.x >> 5;
    int m0    = blockIdx.x * 32 + (wave >> 2) * 16;
    int n0    = blockIdx.y * 64 + (wave & 3) * 16;
    int khalf = (lane & 16) ? 8 : 0;
    int arow  = m0 + (lane & 15);
    int ncol  = n0 + (lane & 15);

    v8f acc = {};
    for (int k0 = 0; k0 < K; k0 += 32) {
        v16h a = load_frag(A, lda, arow, k0, khalf);
        v16h b = load_frag(W, ldw, ncol, k0, khalf);
        acc = __builtin_amdgcn_wmma_f32_16x16x32_f16(
            false, a, false, b, (short)0, acc, false, false);
    }
    float bia = bias ? bias[ncol] : 0.0f;
    int mbase = m0 + ((lane & 16) ? 8 : 0);
#pragma unroll
    for (int r = 0; r < 8; ++r) {
        int m = mbase + r;
        float v = acc[r] + bia;
        if (relu) v = v > 0.0f ? v : 0.0f;
        size_t idx = (size_t)m * N + ncol;
        if (res)  v += res[idx];
        if (outF) outF[idx] = v;
        if (outH) outH[idx] = (_Float16)v;
    }
}

// ---------------- attention: one block per (b, h, row-half of 64) ----------------
// qkv packed [M,384] f16: q cols 0..127, k cols 128..255, v cols 256..383 (col = h*32+hd).
// LDS budget: 33792 + 18432 + 9216 = 61440 B (< 64 KB).
__global__ __launch_bounds__(256) void attn_kernel(
    const _Float16* __restrict__ qkv, _Float16* __restrict__ attn_out) {
    __shared__ float                  sc[64][132];   // fp32 scores (this 64-row half)
    __shared__ __align__(16) _Float16 wh[64][144];   // softmax weights, f16
    __shared__ __align__(16) _Float16 vT[32][144];   // V transposed [hd][s]

    int blk  = blockIdx.x;
    int half = blk & 1;            // which 64-row half of the 128 queries
    int bh   = blk >> 1;
    int b    = bh >> 2, h = bh & 3;
    int r0   = half * 64;

    const _Float16* base = qkv + (size_t)b * TT * 384;
    const _Float16* qb   = base + h * HDIM;          // q
    const _Float16* kb   = base + 128 + h * HDIM;    // k
    const _Float16* vb   = base + 256 + h * HDIM;    // v

    int tid = threadIdx.x, lane = tid & 31, wave = tid >> 5;

    // stage V^T into LDS (coalesced-ish: 4096 halfs)
    for (int i = tid; i < TT * HDIM; i += 256) {
        int s = i >> 5, d = i & 31;
        vT[d][s] = vb[(size_t)s * 384 + d];
    }

    // ---- scores = Q K^T * SCALE, causal mask → LDS fp32 ----
    // waves: 4 row-tiles x 2 col-halves; each wave does 4 WMMAs (K=HD=32, single chunk)
    int khalf = (lane & 16) ? 8 : 0;
    int m0    = (wave >> 1) * 16;          // 0,16,32,48 (local)
    int nbase = (wave & 1) * 64;
    v16h aq = load_frag(qb, 384, r0 + m0 + (lane & 15), 0, khalf);
#pragma unroll
    for (int nt = 0; nt < 4; ++nt) {
        int n0 = nbase + nt * 16;
        v16h bk = load_frag(kb, 384, n0 + (lane & 15), 0, khalf);
        v8f c = {};
        c = __builtin_amdgcn_wmma_f32_16x16x32_f16(
            false, aq, false, bk, (short)0, c, false, false);
        int nc = n0 + (lane & 15);
        int mb = m0 + ((lane & 16) ? 8 : 0);
#pragma unroll
        for (int r = 0; r < 8; ++r) {
            int ml = mb + r;
            float s = c[r] * SCALE;
            if (nc > r0 + ml) s = -__builtin_huge_valf();   // causal mask
            sc[ml][nc] = s;
        }
    }
    __syncthreads();

    // ---- softmax, one thread per row (rows 0..63 of this half) ----
    if (tid < 64) {
        float mx = -__builtin_huge_valf();
#pragma unroll 4
        for (int j = 0; j < TT; ++j) mx = fmaxf(mx, sc[tid][j]);
        float sum = 0.0f;
#pragma unroll 4
        for (int j = 0; j < TT; ++j) {
            float e = __expf(sc[tid][j] - mx);
            sc[tid][j] = e;
            sum += e;
        }
        float inv = 1.0f / sum;
#pragma unroll 4
        for (int j = 0; j < TT; ++j) wh[tid][j] = (_Float16)(sc[tid][j] * inv);
    }
    __syncthreads();

    // ---- attn = W @ V : 64 x 32, waves: 4 row-tiles x 2 col-tiles ----
    int am0 = (wave >> 1) * 16;
    int an0 = (wave & 1) * 16;
    v8f c = {};
#pragma unroll
    for (int kc = 0; kc < 4; ++kc) {
        v16h a = load_frag(&wh[0][0], 144, am0 + (lane & 15), kc * 32, khalf);
        v16h v = load_frag(&vT[0][0], 144, an0 + (lane & 15), kc * 32, khalf);
        c = __builtin_amdgcn_wmma_f32_16x16x32_f16(
            false, a, false, v, (short)0, c, false, false);
    }
    int nc = an0 + (lane & 15);
    int mb = am0 + ((lane & 16) ? 8 : 0);
#pragma unroll
    for (int r = 0; r < 8; ++r) {
        int gm = r0 + mb + r;
        attn_out[(size_t)(b * TT + gm) * DD + h * HDIM + nc] = (_Float16)c[r];
    }
}

// ---------------- host orchestration ----------------
extern "C" void kernel_launch(void* const* d_in, const int* in_sizes, int n_in,
                              void* d_out, int out_size, void* d_ws, size_t ws_size,
                              hipStream_t stream) {
    const float* x     = (const float*)d_in[0];
    const float* ln1_g = (const float*)d_in[1];
    const float* ln1_b = (const float*)d_in[2];
    const float* Wq    = (const float*)d_in[3];
    const float* Wk    = (const float*)d_in[4];
    const float* Wv    = (const float*)d_in[5];
    const float* Wo    = (const float*)d_in[6];
    const float* bo    = (const float*)d_in[7];
    const float* ln2_g = (const float*)d_in[8];
    const float* ln2_b = (const float*)d_in[9];
    const float* W1    = (const float*)d_in[10];
    const float* b1    = (const float*)d_in[11];
    const float* W2    = (const float*)d_in[12];
    const float* b2    = (const float*)d_in[13];
    float* out = (float*)d_out;

    // workspace layout (bytes)
    char* wsb = (char*)d_ws;
    const size_t OFF_W16  = 0;                              // 393216 B of f16 weights
    const size_t OFF_H16  = 524288;                         // LN out, f16 [M,128] 32MB (reused for LN2)
    const size_t OFF_QKV  = OFF_H16 + (size_t)33554432;     // f16 [M,384] 96MB
    const size_t OFF_ATTN = OFF_QKV + (size_t)100663296;    // f16 [M,128] 32MB
    const size_t OFF_X2   = OFF_ATTN + (size_t)33554432;    // fp32 [M,128] 64MB
    const size_t OFF_FF1  = OFF_QKV;                        // f16 [M,512] 128MB (reuses qkv+attn)

    _Float16* w16    = (_Float16*)(wsb + OFF_W16);
    _Float16* wqkv16 = w16;                 // [384,128]
    _Float16* wo16   = w16 + 49152;         // [128,128]
    _Float16* w1_16  = w16 + 65536;         // [512,128]
    _Float16* w2_16  = w16 + 131072;        // [128,512]
    _Float16* h16    = (_Float16*)(wsb + OFF_H16);
    _Float16* qkv16  = (_Float16*)(wsb + OFF_QKV);
    _Float16* attn16 = (_Float16*)(wsb + OFF_ATTN);
    float*    x2     = (float*)   (wsb + OFF_X2);
    _Float16* ff1    = (_Float16*)(wsb + OFF_FF1);

    dim3 blk(256);

    // 1) pack weights to f16
    cvt_weights_kernel<<<768, blk, 0, stream>>>(Wq, Wk, Wv, Wo, W1, W2, w16);

    // 2) LN1: x -> h16
    ln_f16_kernel<<<MROWS / 8, blk, 0, stream>>>(x, ln1_g, ln1_b, h16);

    // 3) fused QKV projection: [M,128] @ [384,128]^T -> qkv16 [M,384]
    wmma_gemm_kernel<<<dim3(MROWS / 32, 6), blk, 0, stream>>>(
        h16, DD, wqkv16, DD, nullptr, nullptr, nullptr, qkv16, 384, DD, 0);

    // 4) attention (scores+softmax+PV), 2 blocks per (b,h)
    attn_kernel<<<BB * HH * 2, blk, 0, stream>>>(qkv16, attn16);

    // 5) out-proj + bias + residual: x2 = x + attn @ Wo^T + bo
    wmma_gemm_kernel<<<dim3(MROWS / 32, 2), blk, 0, stream>>>(
        attn16, DD, wo16, DD, bo, x, x2, nullptr, DD, DD, 0);

    // 6) LN2: x2 -> h16 (reuse)
    ln_f16_kernel<<<MROWS / 8, blk, 0, stream>>>(x2, ln2_g, ln2_b, h16);

    // 7) FF1 + bias + ReLU: [M,128] @ [512,128]^T -> ff1 [M,512] f16
    wmma_gemm_kernel<<<dim3(MROWS / 32, 8), blk, 0, stream>>>(
        h16, DD, w1_16, DD, b1, nullptr, nullptr, ff1, FF, DD, 1);

    // 8) FF2 + bias + residual: out = x2 + ff1 @ W2^T + b2  (fp32)
    wmma_gemm_kernel<<<dim3(MROWS / 32, 2), blk, 0, stream>>>(
        ff1, FF, w2_16, FF, b2, x2, out, nullptr, DD, FF, 0);
}
